// PQN_Craftax_Agent_68289980006781
// MI455X (gfx1250) — compile-verified, matrix operand order checked by reading, not run on hardware
//
#include <hip/hip_runtime.h>
#include <stdint.h>

// ---------------------------------------------------------------------------
// Types for CDNA5 WMMA (gfx1250, wave32)
// ---------------------------------------------------------------------------
typedef __bf16  v16bf __attribute__((ext_vector_type(16)));
typedef float   v8f   __attribute__((ext_vector_type(8)));

#define TB     16384   // T*B
#define OBS    8268
#define OBSP   8288    // OBS padded to multiple of 32
#define HID    512
#define LH     128     // LSTM hidden
#define NACT   43
#define TSTEPS 128
#define BATCH  128

#define LDSK   40      // LDS row stride in bf16 elements (80 B, 16B-aligned chunks)

// ---------------------------------------------------------------------------
// Async copy: 16 bytes per lane, global -> LDS, tracked by ASYNCcnt.
// ---------------------------------------------------------------------------
__device__ __forceinline__ void async_cp16(unsigned lds_off, const __bf16* gptr) {
    uint64_t g = (uint64_t)(uintptr_t)gptr;
    asm volatile("global_load_async_to_lds_b128 %0, %1, off"
                 :: "v"(lds_off), "v"(g) : "memory");
}
__device__ __forceinline__ void async_wait0() {
    asm volatile("s_wait_asynccnt 0" ::: "memory");
}
__device__ __forceinline__ unsigned lds_addr(const void* p) {
    return (unsigned)(uintptr_t)p;   // low 32 bits of generic shared addr = LDS offset
}

// ---------------------------------------------------------------------------
// Fragment loader: 16-bit A/B fragment per documented CDNA5 layout.
// lanes 0-15: elems 0..7 = K 0..7,  elems 8..15 = K 16..23  (cb = 0)
// lanes16-31: elems 0..7 = K 8..15, elems 8..15 = K 24..31  (cb = 8)
// ---------------------------------------------------------------------------
__device__ __forceinline__ v16bf ldfrag(const __bf16* row, int cb) {
    v16bf v;
#pragma unroll
    for (int e = 0; e < 8; ++e) {
        v[e]     = row[cb + e];
        v[e + 8] = row[cb + 16 + e];
    }
    return v;
}

__device__ __forceinline__ float sigm(float x) { return 1.0f / (1.0f + __expf(-x)); }

// ---------------------------------------------------------------------------
// Weight prep: fp32 -> bf16, transposed into [Npad, Kp] row-major (K fastest),
// zero-padded so GEMM staging needs no bounds checks and no NaN risk.
// SRC_NK: source already [N,K] row-major; else source is [K,N].
// ---------------------------------------------------------------------------
template <bool SRC_NK>
__global__ __launch_bounds__(256) void prep_weight(
    const float* __restrict__ W, __bf16* __restrict__ WT,
    int K, int N, int Kp, int Npad)
{
    int i = blockIdx.x * 256 + threadIdx.x;
    if (i >= Npad * Kp) return;
    int n = i / Kp, k = i - n * Kp;
    float v = 0.0f;
    if (n < N && k < K)
        v = SRC_NK ? W[(size_t)n * K + k] : W[(size_t)k * N + n];
    WT[i] = (__bf16)v;
}

// ---------------------------------------------------------------------------
// GEMM: C[M,ldc] = A[M,K] * Bt^T (+bias +bias2), bf16 WMMA, f32 accum.
// Bt: pre-converted bf16 weights [Npad, Kp] (zero padded). A: fp32 (A_F32,
// converted through VGPRs) or bf16 (pure async global->LDS copy).
// 256 thr = 8 waves; block tile 128x128; wave tile 32x64; double-buffered LDS.
// ---------------------------------------------------------------------------
template <bool A_F32>
__global__ __launch_bounds__(256) void gemm_wmma(
    const void* __restrict__ Ap, const __bf16* __restrict__ Bt,
    float* __restrict__ C, const float* __restrict__ bias,
    const float* __restrict__ bias2, int M, int N, int K, int Kp, int ldc)
{
    __shared__ __bf16 As[2][128][LDSK];   // 20 KB
    __shared__ __bf16 Bs[2][128][LDSK];   // 20 KB

    const float*  Af = (const float*)Ap;
    const __bf16* Ab = (const __bf16*)Ap;

    const int tid  = threadIdx.x;
    const int lane = tid & 31;
    const int wave = tid >> 5;
    const int wm   = (wave & 3) * 32;
    const int wn   = (wave >> 2) * 64;
    const int m0   = blockIdx.y * 128;
    const int n0   = blockIdx.x * 128;

    const int srow = tid >> 1;            // fp32-A staging: row 0..127
    const int kseg = (tid & 1) * 16;      // fp32-A staging: k segment 0/16
    const int lr   = lane & 15;
    const int cb   = (lane < 16) ? 0 : 8;
    const int ktiles = Kp >> 5;

    // ---- staging helpers ----
    auto issueB = [&](int kt, int buf) {   // 2 async b128 per thread
#pragma unroll
        for (int i = 0; i < 2; ++i) {
            int c  = tid * 2 + i;
            int n  = c >> 2;
            int sg = (c & 3) * 8;
            async_cp16(lds_addr(&Bs[buf][n][sg]),
                       Bt + (size_t)(n0 + n) * Kp + (kt << 5) + sg);
        }
    };
    auto issueA_bf16 = [&](int kt, int buf) {
#pragma unroll
        for (int i = 0; i < 2; ++i) {
            int c  = tid * 2 + i;
            int m  = c >> 2;
            int sg = (c & 3) * 8;
            async_cp16(lds_addr(&As[buf][m][sg]),
                       Ab + (size_t)(m0 + m) * K + (kt << 5) + sg);
        }
    };
    auto loadA_f32 = [&](int kt, float* ar) {
        const int k0 = kt << 5;
        const size_t base = (size_t)(m0 + srow) * K;
        if (k0 + 32 <= K) {
            const float4* p = (const float4*)(Af + base + k0 + kseg);
            __builtin_prefetch(Af + base + k0 + 128, 0, 1);
            float4 f0 = p[0], f1 = p[1], f2 = p[2], f3 = p[3];
            ar[0]=f0.x; ar[1]=f0.y; ar[2]=f0.z; ar[3]=f0.w;
            ar[4]=f1.x; ar[5]=f1.y; ar[6]=f1.z; ar[7]=f1.w;
            ar[8]=f2.x; ar[9]=f2.y; ar[10]=f2.z; ar[11]=f2.w;
            ar[12]=f3.x; ar[13]=f3.y; ar[14]=f3.z; ar[15]=f3.w;
        } else {
#pragma unroll
            for (int e = 0; e < 16; ++e) {
                int k = k0 + kseg + e;
                ar[e] = (k < K) ? Af[base + k] : 0.0f;   // zero x K padded B
            }
        }
    };
    auto storeA_f32 = [&](int buf, const float* ar) {
        __bf16 tmp[16];
#pragma unroll
        for (int e = 0; e < 16; ++e) tmp[e] = (__bf16)ar[e];
        uint4* dst = (uint4*)&As[buf][srow][kseg];
        dst[0] = ((const uint4*)tmp)[0];
        dst[1] = ((const uint4*)tmp)[1];
    };

    v8f acc[2][4];
#pragma unroll
    for (int mt = 0; mt < 2; ++mt)
#pragma unroll
        for (int nt = 0; nt < 4; ++nt)
#pragma unroll
            for (int r = 0; r < 8; ++r) acc[mt][nt][r] = 0.0f;

    // ---- prologue: tile 0 ----
    float ar[16];
    issueB(0, 0);
    if constexpr (A_F32) { loadA_f32(0, ar); storeA_f32(0, ar); }
    else                 { issueA_bf16(0, 0); }
    async_wait0();
    __syncthreads();

    // ---- main loop: compute buf, async-fill buf^1 ----
    for (int kt = 0; kt < ktiles; ++kt) {
        const int buf  = kt & 1;
        const int nbuf = buf ^ 1;
        const bool more = (kt + 1) < ktiles;
        if (more) {
            issueB(kt + 1, nbuf);
            if constexpr (A_F32) loadA_f32(kt + 1, ar);
            else                 issueA_bf16(kt + 1, nbuf);
        }

        v16bf a[2], b[4];
#pragma unroll
        for (int mt = 0; mt < 2; ++mt) a[mt] = ldfrag(&As[buf][wm + mt * 16 + lr][0], cb);
#pragma unroll
        for (int nt = 0; nt < 4; ++nt) b[nt] = ldfrag(&Bs[buf][wn + nt * 16 + lr][0], cb);
#pragma unroll
        for (int mt = 0; mt < 2; ++mt)
#pragma unroll
            for (int nt = 0; nt < 4; ++nt)
                acc[mt][nt] = __builtin_amdgcn_wmma_f32_16x16x32_bf16(
                    false, a[mt], false, b[nt], (short)0, acc[mt][nt], false, false);

        if (more) {
            if constexpr (A_F32) storeA_f32(nbuf, ar);
            async_wait0();
            __syncthreads();
        }
    }

    // ---- epilogue: bias + fp32 store ----
#pragma unroll
    for (int mt = 0; mt < 2; ++mt)
#pragma unroll
        for (int nt = 0; nt < 4; ++nt)
#pragma unroll
            for (int r = 0; r < 8; ++r) {
                int gm = m0 + wm + mt * 16 + r + ((lane < 16) ? 0 : 8);
                int gn = n0 + wn + nt * 16 + (lane & 15);
                if (gn < N) {
                    float bv = acc[mt][nt][r];
                    if (bias)  bv += bias[gn];
                    if (bias2) bv += bias2[gn];
                    C[(size_t)gm * ldc + gn] = bv;
                }
            }
}

// ---------------------------------------------------------------------------
// Row LayerNorm + ReLU, fp32 in -> bf16 out. One-pass sum/sumsq, wave32
// shuffle reductions. One block (128 threads = 4 waves) per row.
// ---------------------------------------------------------------------------
template <int W>
__global__ __launch_bounds__(128) void ln_relu(
    const float* __restrict__ in, const float* __restrict__ g,
    const float* __restrict__ b, __bf16* __restrict__ out)
{
    constexpr int E = W / 128;
    const int row = blockIdx.x;
    const int tid = threadIdx.x;

    float x[E], s = 0.0f, s2 = 0.0f;
#pragma unroll
    for (int e = 0; e < E; ++e) {
        x[e] = in[(size_t)row * W + tid + e * 128];
        s  += x[e];
        s2 += x[e] * x[e];
    }
#pragma unroll
    for (int off = 16; off > 0; off >>= 1) {
        s  += __shfl_xor(s,  off);
        s2 += __shfl_xor(s2, off);
    }
    __shared__ float red[2][4];
    if ((tid & 31) == 0) { red[0][tid >> 5] = s; red[1][tid >> 5] = s2; }
    __syncthreads();
    s  = red[0][0] + red[0][1] + red[0][2] + red[0][3];
    s2 = red[1][0] + red[1][1] + red[1][2] + red[1][3];

    const float mean = s * (1.0f / W);
    const float var  = s2 * (1.0f / W) - mean * mean;
    const float inv  = rsqrtf(var + 1e-5f);

#pragma unroll
    for (int e = 0; e < E; ++e) {
        int col = tid + e * 128;
        float y = (x[e] - mean) * inv * g[col] + b[col];
        out[(size_t)row * W + col] = (__bf16)fmaxf(y, 0.0f);
    }
}

// ---------------------------------------------------------------------------
// Init: reset grid barrier, seed h ping-pong buffer 0 with bf16(h0).
// ---------------------------------------------------------------------------
__global__ void init_state(const float* __restrict__ h0, __bf16* __restrict__ hbuf,
                           int* __restrict__ barrier)
{
    int i = blockIdx.x * blockDim.x + threadIdx.x;
    if (i == 0) barrier[0] = 0;
    if (i < BATCH * LH) hbuf[i] = (__bf16)h0[i];
}

// ---------------------------------------------------------------------------
// Persistent LSTM scan: 8 WGPs x 256 threads. Each WGP owns 16 hidden cols j
// plus their gate replicas (j, j+128, j+256, j+384); each wave owns a 16-row
// batch tile. After 4 gate WMMAs, i/f/g/o for a (b,j) element are in the SAME
// lane/slot -> lane-local cell update; c lives in VGPRs across all 128 steps.
// Cross-WGP sync: release/acquire atomic barrier on double-buffered h.
// ---------------------------------------------------------------------------
__global__ __launch_bounds__(256) void lstm_scan(
    const float* __restrict__ xg, const int* __restrict__ done,
    const float* __restrict__ c0, const float* __restrict__ w_hh,
    __bf16* __restrict__ hbuf, __bf16* __restrict__ hs,
    float* __restrict__ hT, float* __restrict__ cT, int* __restrict__ barrier)
{
    __shared__ __bf16 As[BATCH][LH];      // masked h(t-1), 32 KB
    __shared__ __bf16 Bsh[4][16][LH];     // w_hh^T strips [gate][j][k], 16 KB

    const int tid  = threadIdx.x;
    const int lane = tid & 31;
    const int wave = tid >> 5;
    const int jbase = blockIdx.x * 16;
    const int lr = lane & 15;
    const int cb = (lane < 16) ? 0 : 8;
    const int j  = jbase + lr;

    for (int i = tid; i < 4 * 16 * LH; i += 256) {
        int k  = i & 127;
        int jj = (i >> 7) & 15;
        int gg = i >> 11;
        Bsh[gg][jj][k] = (__bf16)w_hh[(size_t)(gg * 128 + jbase + jj) * LH + k];
    }

    int brow[8];
    float c[8], hlast[8];
#pragma unroll
    for (int r = 0; r < 8; ++r) {
        brow[r]  = wave * 16 + r + ((lane < 16) ? 0 : 8);
        c[r]     = c0[(size_t)brow[r] * LH + j];
        hlast[r] = 0.0f;
    }
    __syncthreads();

    for (int t = 0; t < TSTEPS; ++t) {
        const __bf16* hin  = hbuf + (size_t)(t & 1) * (BATCH * LH);
        __bf16*       hout = hbuf + (size_t)((t + 1) & 1) * (BATCH * LH);

        for (int i = tid; i < BATCH * LH; i += 256) {
            int bb = i >> 7;
            As[bb][i & 127] = done[t * BATCH + bb] ? (__bf16)0.0f : hin[i];
        }
        __syncthreads();

        v8f acc[4];
#pragma unroll
        for (int gg = 0; gg < 4; ++gg)
#pragma unroll
            for (int r = 0; r < 8; ++r) acc[gg][r] = 0.0f;

#pragma unroll
        for (int kt = 0; kt < 4; ++kt) {
            v16bf a = ldfrag(&As[wave * 16 + lr][kt * 32], cb);
#pragma unroll
            for (int gg = 0; gg < 4; ++gg) {
                v16bf bb = ldfrag(&Bsh[gg][lr][kt * 32], cb);
                acc[gg] = __builtin_amdgcn_wmma_f32_16x16x32_bf16(
                    false, a, false, bb, (short)0, acc[gg], false, false);
            }
        }

#pragma unroll
        for (int r = 0; r < 8; ++r) {
            const int b = brow[r];
            const size_t xb = ((size_t)t * BATCH + b) * (4 * LH);
            float iv = acc[0][r] + xg[xb + 0 * LH + j];
            float fv = acc[1][r] + xg[xb + 1 * LH + j];
            float gv = acc[2][r] + xg[xb + 2 * LH + j];
            float ov = acc[3][r] + xg[xb + 3 * LH + j];
            float cc = done[t * BATCH + b] ? 0.0f : c[r];
            float cn = sigm(fv) * cc + sigm(iv) * tanhf(gv);
            float hn = sigm(ov) * tanhf(cn);
            c[r] = cn;
            hlast[r] = hn;
            __bf16 hb = (__bf16)hn;
            hout[(size_t)b * LH + j] = hb;
            hs[((size_t)t * BATCH + b) * LH + j] = hb;
        }

        __syncthreads();
        __threadfence();
        if (tid == 0) {
            __hip_atomic_fetch_add(barrier, 1, __ATOMIC_RELEASE, __HIP_MEMORY_SCOPE_AGENT);
            while (__hip_atomic_load(barrier, __ATOMIC_ACQUIRE, __HIP_MEMORY_SCOPE_AGENT)
                   < 8 * (t + 1)) { }
        }
        __syncthreads();
    }

#pragma unroll
    for (int r = 0; r < 8; ++r) {
        hT[(size_t)brow[r] * LH + j] = hlast[r];
        cT[(size_t)brow[r] * LH + j] = c[r];
    }
}

// ---------------------------------------------------------------------------
// Host orchestration
// ---------------------------------------------------------------------------
extern "C" void kernel_launch(void* const* d_in, const int* in_sizes, int n_in,
                              void* d_out, int out_size, void* d_ws, size_t ws_size,
                              hipStream_t stream)
{
    const float* x    = (const float*)d_in[0];
    const int*   done = (const int*)  d_in[1];
    const float* h0   = (const float*)d_in[2];
    const float* c0   = (const float*)d_in[3];
    const float* w_in = (const float*)d_in[4];
    const float* b_in = (const float*)d_in[5];
    const float* w_h  = (const float*)d_in[6];
    const float* b_h  = (const float*)d_in[7];
    const float* w_out= (const float*)d_in[8];
    const float* b_out= (const float*)d_in[9];
    const float* ln_g = (const float*)d_in[10];
    const float* ln_b = (const float*)d_in[11];
    const float* w_ih = (const float*)d_in[12];
    const float* w_hh = (const float*)d_in[13];
    const float* b_ih = (const float*)d_in[14];
    const float* b_hh = (const float*)d_in[15];
    const float* w_p  = (const float*)d_in[16];
    const float* b_p  = (const float*)d_in[17];
    const float* ln_pg= (const float*)d_in[18];
    const float* ln_pb= (const float*)d_in[19];
    const float* w_q  = (const float*)d_in[20];
    const float* b_q  = (const float*)d_in[21];

    // workspace layout
    char* ws = (char*)d_ws;
    float*  P     = (float*) (ws + 0);            // 16384*512 fp32 preact
    __bf16* Abuf  = (__bf16*)(ws + 33554432);     // 16384*512 bf16 act
    float*  XG    = (float*) (ws + 50331648);     // 16384*512 fp32 xg
    __bf16* HBUF  = (__bf16*)(ws + 83886080);     // 2 x 128*128 bf16 h ping-pong
    __bf16* HS    = (__bf16*)(ws + 83951616);     // 128*128*128 bf16 hs
    __bf16* POST  = (__bf16*)(ws + 88145920);     // 16384*128 bf16 post
    int*    BAR   = (int*)   (ws + 92340224);
    __bf16* WTIN  = (__bf16*)(ws + 92340480);     // [512, 8288]
    __bf16* WTH0  = (__bf16*)(ws + 100827392);    // [512, 512]
    __bf16* WTH1  = (__bf16*)(ws + 101351680);
    __bf16* WTOUT = (__bf16*)(ws + 101875968);
    __bf16* WTIH  = (__bf16*)(ws + 102400256);
    __bf16* WTP   = (__bf16*)(ws + 102924544);    // [128, 128]
    __bf16* WTQ   = (__bf16*)(ws + 102957312);    // [128, 128] (rows 43.. zero)

    float* qvals = (float*)d_out;                 // [16384,43]
    float* hT    = qvals + (size_t)TB * NACT;
    float* cT    = hT + BATCH * LH;

    const dim3 blk256(256), blk128(128);
    const dim3 gEnc(HID / 128, TB / 128);         // (4,128)
    const dim3 gPost(1, TB / 128);                // (1,128)

    // ---- one-time per launch: weight prep (bf16, transposed, zero-padded) ----
    prep_weight<false><<<dim3((HID * OBSP + 255) / 256), blk256, 0, stream>>>(w_in,  WTIN,  OBS, HID, OBSP, HID);
    prep_weight<false><<<dim3((HID * HID  + 255) / 256), blk256, 0, stream>>>(w_h,              WTH0, HID, HID, HID, HID);
    prep_weight<false><<<dim3((HID * HID  + 255) / 256), blk256, 0, stream>>>(w_h + HID * HID,  WTH1, HID, HID, HID, HID);
    prep_weight<false><<<dim3((HID * HID  + 255) / 256), blk256, 0, stream>>>(w_out, WTOUT, HID, HID, HID, HID);
    prep_weight<true ><<<dim3((HID * HID  + 255) / 256), blk256, 0, stream>>>(w_ih,  WTIH,  HID, HID, HID, HID);
    prep_weight<false><<<dim3((LH * LH    + 255) / 256), blk256, 0, stream>>>(w_p,   WTP,   LH,  LH,  LH,  LH);
    prep_weight<false><<<dim3((LH * LH    + 255) / 256), blk256, 0, stream>>>(w_q,   WTQ,   LH,  NACT, LH, LH);

    init_state<<<dim3(64), blk256, 0, stream>>>(h0, HBUF, BAR);

    // ---- encoder: Linear -> LN -> ReLU (x4) ----
    gemm_wmma<true ><<<gEnc, blk256, 0, stream>>>(x,    WTIN,  P, b_in,      nullptr, TB, HID, OBS, OBSP, HID);
    ln_relu<HID><<<dim3(TB), blk128, 0, stream>>>(P, ln_g + 0 * HID, ln_b + 0 * HID, Abuf);
    gemm_wmma<false><<<gEnc, blk256, 0, stream>>>(Abuf, WTH0,  P, b_h,       nullptr, TB, HID, HID, HID, HID);
    ln_relu<HID><<<dim3(TB), blk128, 0, stream>>>(P, ln_g + 1 * HID, ln_b + 1 * HID, Abuf);
    gemm_wmma<false><<<gEnc, blk256, 0, stream>>>(Abuf, WTH1,  P, b_h + HID, nullptr, TB, HID, HID, HID, HID);
    ln_relu<HID><<<dim3(TB), blk128, 0, stream>>>(P, ln_g + 2 * HID, ln_b + 2 * HID, Abuf);
    gemm_wmma<false><<<gEnc, blk256, 0, stream>>>(Abuf, WTOUT, P, b_out,     nullptr, TB, HID, HID, HID, HID);
    ln_relu<HID><<<dim3(TB), blk128, 0, stream>>>(P, ln_g + 3 * HID, ln_b + 3 * HID, Abuf);

    // ---- xg = seq @ w_ih^T + b_ih + b_hh ----
    gemm_wmma<false><<<gEnc, blk256, 0, stream>>>(Abuf, WTIH, XG, b_ih, b_hh, TB, 4 * LH, HID, HID, 4 * LH);

    // ---- sequential LSTM scan (persistent 8-WGP WMMA kernel) ----
    lstm_scan<<<dim3(8), blk256, 0, stream>>>(XG, done, c0, w_hh, HBUF, HS, hT, cT, BAR);

    // ---- post: Linear -> LN -> ReLU, then qvals ----
    gemm_wmma<false><<<gPost, blk256, 0, stream>>>(HS, WTP, P, b_p, nullptr, TB, LH, LH, LH, LH);
    ln_relu<LH><<<dim3(TB), blk128, 0, stream>>>(P, ln_pg, ln_pb, POST);
    gemm_wmma<false><<<gPost, blk256, 0, stream>>>(POST, WTQ, qvals, b_q, nullptr, TB, NACT, LH, LH, NACT);
}
